// CIFlow_20160576487452
// MI455X (gfx1250) — compile-verified, matrix operand order checked by reading, not run on hardware
//
#include <hip/hip_runtime.h>
#include <hip/hip_bf16.h>
#include <math.h>

// ---------------- problem constants (from reference) ----------------
constexpr int Bg   = 256;          // graphs / batch
constexpr int Mq   = 512;          // nodes per graph in Q/E
constexpr int KC   = 10;           // clusters
constexpr int Dd   = 64;           // feature dim
constexpr int Nn   = 131072;       // nodes (S/H)
constexpr int NNZ  = 2097152;      // sparse entries
constexpr int NQ   = Bg * Mq;      // 131072 flattened (b,m) rows
constexpr int NPG  = Nn / Bg;      // 512 nodes per graph (batch = n >> 9)

// ---------------- ws layout (float offsets) ----------------
constexpr int OFF_SC   = 0;                          // scalars
constexpr int OFF_QMIN = 16;
constexpr int OFF_QCNT = 32;
constexpr int OFF_DEN  = 48;                         // denom[B*K]
constexpr int OFF_SSB  = OFF_DEN + Bg * KC;          // 2608: ss_buf[16*16]
constexpr int OFF_PRO  = OFF_SSB + 256;              // 2864: proto_buf[16*64]
constexpr int OFF_CG   = OFF_PRO + 16 * 64;          // 3888: c_g[B]
constexpr int OFF_SN   = OFF_CG + Bg;                // 4144: S_n[N*K]

typedef __attribute__((ext_vector_type(16))) _Float16 v16h;
typedef __attribute__((ext_vector_type(8)))  float    v8f;
typedef __attribute__((ext_vector_type(2)))  float    v2f;

// deterministic per-node uniform in [0,1) (splitmix64-style hash)
__device__ inline float rng01(unsigned int n) {
    unsigned long long z = (unsigned long long)(n + 1u) * 0x9E3779B97F4A7C15ull + 0x42ull;
    z ^= z >> 30; z *= 0xBF58476D1CE4E5B9ull;
    z ^= z >> 27; z *= 0x94D049BB133111EBull;
    z ^= z >> 31;
    return (float)(z >> 40) * (1.0f / 16777216.0f);
}

// ---------------- init accumulators ----------------
__global__ void k_init(float* ws) {
    int i = blockIdx.x * blockDim.x + threadIdx.x;
    if (i < 16)  ws[OFF_SC + i] = 0.0f;
    if (i < 16)  ws[OFF_QMIN + i] = 1e30f;
    if (i < 16)  ws[OFF_QCNT + i] = 0.0f;
    if (i < 256) ws[OFF_SSB + i] = 0.0f;
    if (i < 16 * 64) ws[OFF_PRO + i] = 0.0f;
}

// ---------------- loss_1 / loss_2 (cross entropies) ----------------
__global__ void k_ce(const float* __restrict__ p1, const float* __restrict__ p2,
                     const int* __restrict__ labels, const unsigned char* __restrict__ ind,
                     float* ws) {
    __shared__ float r1[256], r2[256], r3[256];
    int b = threadIdx.x;
    float a = p1[b * 2 + 0], c = p1[b * 2 + 1];
    float mx = fmaxf(a, c);
    float lse = mx + logf(expf(a - mx) + expf(c - mx));
    int l = labels[b];
    float ce1 = lse - (l == 0 ? a : c);
    float a2 = p2[b * 2 + 0], c2 = p2[b * 2 + 1];
    float mx2 = fmaxf(a2, c2);
    float lse2 = mx2 + logf(expf(a2 - mx2) + expf(c2 - mx2));
    float ce2 = lse2 - (l == 0 ? a2 : c2);
    float mk = ind[b] ? 1.0f : 0.0f;
    r1[b] = ce1; r2[b] = mk * ce2; r3[b] = mk;
    __syncthreads();
    for (int o = 128; o > 0; o >>= 1) {
        if (b < o) { r1[b] += r1[b + o]; r2[b] += r2[b + o]; r3[b] += r3[b + o]; }
        __syncthreads();
    }
    if (b == 0) {
        ws[OFF_SC + 0] = r1[0] / (float)Bg;
        float np = r3[0];
        ws[OFF_SC + 1] = 0.1f * (np > 0.0f ? r2[0] / fmaxf(np, 1.0f) : 0.0f);
    }
}

// ---------------- per-graph column L2 norms of S ----------------
__global__ void k_denom(const float* __restrict__ S, float* __restrict__ den) {
    __shared__ float acc[KC];
    if (threadIdx.x < KC) acc[threadIdx.x] = 0.0f;
    __syncthreads();
    int b = blockIdx.x;
    float la[KC];
#pragma unroll
    for (int k = 0; k < KC; ++k) la[k] = 0.0f;
    for (int r = threadIdx.x; r < NPG; r += blockDim.x) {
        const float* row = S + (size_t)(b * NPG + r) * KC;
#pragma unroll
        for (int k = 0; k < KC; ++k) la[k] += row[k] * row[k];
    }
#pragma unroll
    for (int k = 0; k < KC; ++k) atomicAdd(&acc[k], la[k]);
    __syncthreads();
    if (threadIdx.x < KC)
        den[b * KC + threadIdx.x] = sqrtf(acc[threadIdx.x]) + 1e-5f;
}

// ---------------- S_n = S / denom[batch] ----------------
__global__ void k_snorm(const float* __restrict__ S, const float* __restrict__ den,
                        float* __restrict__ Sn) {
    int n = blockIdx.x * blockDim.x + threadIdx.x;
    if (n >= Nn) return;
    int b = n >> 9; // batch = n // 512
#pragma unroll
    for (int k = 0; k < KC; ++k)
        Sn[(size_t)n * KC + k] = S[(size_t)n * KC + k] / den[b * KC + k];
}

// ---------------- loss_sp = sum v * <S_n[r], S_n[c]> ----------------
__global__ void k_losssp(const float* __restrict__ Sn, const int* __restrict__ Lr,
                         const int* __restrict__ Lc, const float* __restrict__ Lv,
                         float* ws) {
    __shared__ float red[256];
    int i = blockIdx.x * blockDim.x + threadIdx.x;
    float v = 0.0f;
    if (i < NNZ) {
        if (i + 16384 < NNZ) {  // prefetch the index/value streams ahead
            __builtin_prefetch(&Lr[i + 16384], 0, 0);
            __builtin_prefetch(&Lc[i + 16384], 0, 0);
            __builtin_prefetch(&Lv[i + 16384], 0, 0);
        }
        const float* ar = Sn + (size_t)Lr[i] * KC;
        const float* bc = Sn + (size_t)Lc[i] * KC;
        float d = 0.0f;
#pragma unroll
        for (int k = 0; k < KC; ++k) d += ar[k] * bc[k];
        v = Lv[i] * d;
    }
    red[threadIdx.x] = v;
    __syncthreads();
    for (int o = 128; o > 0; o >>= 1) {
        if (threadIdx.x < o) red[threadIdx.x] += red[threadIdx.x + o];
        __syncthreads();
    }
    if (threadIdx.x == 0) atomicAdd(&ws[OFF_SC + 2], red[0]);
}

// ---------------- ss = S_n^T S_n via V_WMMA_F32_16X16X4_F32 ----------------
// A (16x4) lane: m=lane&15, h=lane>>4 -> {Sn[n0+2h][m], Sn[n0+2h+1][m]}
// B (4x16) lane: identical values (A==B per lane here). Branchless masked loads.
__global__ void k_ss_wmma(const float* __restrict__ Sn, float* __restrict__ ssb) {
    int lane = threadIdx.x & 31;
    int wid  = (blockIdx.x * blockDim.x + threadIdx.x) >> 5;
    int nW   = (gridDim.x * blockDim.x) >> 5;
    int h = lane >> 4, m = lane & 15;
    int   mc  = (m < KC) ? m : (KC - 1);
    float msk = (m < KC) ? 1.0f : 0.0f;
    v8f acc = {};
    for (int c = wid; c < Nn / 4; c += nW) {
        int n0 = c * 4;
        v2f A;
        A.x = Sn[(size_t)(n0 + 2 * h) * KC + mc] * msk;
        A.y = Sn[(size_t)(n0 + 2 * h + 1) * KC + mc] * msk;
        acc = __builtin_amdgcn_wmma_f32_16x16x4_f32(
            false, A, false, A, (short)0, acc, false, false);
    }
    int j = lane & 15;
#pragma unroll
    for (int r = 0; r < 8; ++r) {
        int mo = r + 8 * h;
        atomicAdd(&ssb[mo * 16 + j], acc[r]);
    }
}

// ---------------- proto = sum_n Q[n,:]^T E[n,:] via v_wmma_f32_16x16x32_f16 ----
// E tile (32 rows x 64 f32 = 8KB, linear) is staged global->LDS with async-LDS
// copies (ASYNCcnt path), then per-lane WMMA-layout gathers read from LDS.
__global__ void __launch_bounds__(128) k_proto_wmma(const float* __restrict__ Q,
                                                    const float* __restrict__ E,
                                                    float* __restrict__ pro) {
    __shared__ float eTile[4][32 * Dd];   // one 8KB tile per wave
    int lane = threadIdx.x & 31;
    int w    = threadIdx.x >> 5;          // wave in block
    int wid  = (blockIdx.x * blockDim.x + threadIdx.x) >> 5;
    int nW   = (gridDim.x * blockDim.x) >> 5;
    int h = lane >> 4, m = lane & 15, j = lane & 15;
    int   mc  = (m < KC) ? m : (KC - 1);
    float msk = (m < KC) ? 1.0f : 0.0f;
    unsigned ldsBase = (unsigned)(uintptr_t)&eTile[w][0]; // low 32 bits = LDS offset
    v8f acc0 = {}, acc1 = {}, acc2 = {}, acc3 = {};
    for (int c = wid; c < NQ / 32; c += nW) {
        int n0 = c * 32;
        // ---- async stage: 512 float4 = 16 b128 copies per lane ----
        unsigned gByte0 = (unsigned)n0 * (unsigned)(Dd * 4); // byte offset of tile
#pragma unroll
        for (int t = 0; t < 16; ++t) {
            int f4 = t * 32 + lane;                       // float4 index in tile
            unsigned lofs = ldsBase + (unsigned)f4 * 16u;
            unsigned gofs = gByte0 + (unsigned)f4 * 16u;
            asm volatile("global_load_async_to_lds_b128 %0, %1, %2"
                         :: "v"(lofs), "v"(gofs), "s"(E) : "memory");
        }
        // ---- A operand from Q (branchless masked scalar loads) ----
        v16h Ah;
#pragma unroll
        for (int v = 0; v < 8; ++v) {
            int k0 = ((v < 4) ? 0 : 16) + 8 * h + 2 * (v & 3);
            float q0 = Q[(size_t)(n0 + k0) * KC + mc] * msk;
            float q1 = Q[(size_t)(n0 + k0 + 1) * KC + mc] * msk;
            Ah[2 * v]     = (_Float16)q0;
            Ah[2 * v + 1] = (_Float16)q1;
        }
        asm volatile("s_wait_asynccnt 0" ::: "memory");   // tile resident in LDS
        // ---- 4 D-tiles of B from LDS, 4 WMMAs ----
#pragma unroll
        for (int t = 0; t < 4; ++t) {
            v16h Bh;
#pragma unroll
            for (int v = 0; v < 8; ++v) {
                int k0 = ((v < 4) ? 0 : 16) + 8 * h + 2 * (v & 3);
                Bh[2 * v]     = (_Float16)eTile[w][k0 * Dd + t * 16 + j];
                Bh[2 * v + 1] = (_Float16)eTile[w][(k0 + 1) * Dd + t * 16 + j];
            }
            if (t == 0) acc0 = __builtin_amdgcn_wmma_f32_16x16x32_f16(false, Ah, false, Bh, (short)0, acc0, false, false);
            if (t == 1) acc1 = __builtin_amdgcn_wmma_f32_16x16x32_f16(false, Ah, false, Bh, (short)0, acc1, false, false);
            if (t == 2) acc2 = __builtin_amdgcn_wmma_f32_16x16x32_f16(false, Ah, false, Bh, (short)0, acc2, false, false);
            if (t == 3) acc3 = __builtin_amdgcn_wmma_f32_16x16x32_f16(false, Ah, false, Bh, (short)0, acc3, false, false);
        }
    }
#pragma unroll
    for (int r = 0; r < 8; ++r) {
        int mo = r + 8 * h;
        atomicAdd(&pro[mo * Dd +  0 + j], acc0[r]);
        atomicAdd(&pro[mo * Dd + 16 + j], acc1[r]);
        atomicAdd(&pro[mo * Dd + 32 + j], acc2[r]);
        atomicAdd(&pro[mo * Dd + 48 + j], acc3[r]);
    }
}

// ---------------- per-k column sums of Q and column mins of (1-Q) ----------------
__global__ void k_qstats(const float* __restrict__ Q, float* ws) {
    __shared__ float ssum[KC];
    __shared__ unsigned int smin[KC];
    if (threadIdx.x < KC) { ssum[threadIdx.x] = 0.0f; smin[threadIdx.x] = __float_as_uint(1e30f); }
    __syncthreads();
    float ls[KC], lm[KC];
#pragma unroll
    for (int k = 0; k < KC; ++k) { ls[k] = 0.0f; lm[k] = 1e30f; }
    int stride = gridDim.x * blockDim.x;
    for (int n = blockIdx.x * blockDim.x + threadIdx.x; n < NQ; n += stride) {
        const float* row = Q + (size_t)n * KC;
#pragma unroll
        for (int k = 0; k < KC; ++k) {
            float q = row[k];
            ls[k] += q;
            lm[k] = fminf(lm[k], 1.0f - q);
        }
    }
#pragma unroll
    for (int k = 0; k < KC; ++k) {
        atomicAdd(&ssum[k], ls[k]);
        atomicMin(&smin[k], __float_as_uint(lm[k])); // positive -> bit-monotonic
    }
    __syncthreads();
    if (threadIdx.x < KC) {
        atomicAdd(&ws[OFF_QCNT + threadIdx.x], ssum[threadIdx.x]);
        atomicMin((unsigned int*)&ws[OFF_QMIN + threadIdx.x], smin[threadIdx.x]);
    }
}

// ---------------- feature loss: per-graph hard clustering ----------------
__global__ void k_feature(const float* __restrict__ S, const float* __restrict__ H,
                          float* ws) {
    __shared__ int   asg[NPG];
    __shared__ float sums[KC * Dd];
    __shared__ float cnt[KC], fds[KC];
    __shared__ float cgacc;
    int tid = threadIdx.x, b = blockIdx.x;
    for (int i = tid; i < KC * Dd; i += blockDim.x) sums[i] = 0.0f;
    if (tid < KC) { cnt[tid] = 0.0f; fds[tid] = 0.0f; }
    if (tid == 0) cgacc = 0.0f;
    __syncthreads();
    // pass 1: sample hard assignment (inverse-CDF on softmax row), counts + sums
    for (int r = tid; r < NPG; r += blockDim.x) {
        int n = b * NPG + r;
        float u = rng01((unsigned int)n);
        const float* srow = S + (size_t)n * KC;
        int k = KC - 1;
        float a = 0.0f;
        for (int kk = 0; kk < KC; ++kk) { a += srow[kk]; if (u < a) { k = kk; break; } }
        asg[r] = k;
        atomicAdd(&cnt[k], 1.0f);
        const float* hrow = H + (size_t)n * Dd;
#pragma unroll
        for (int d = 0; d < Dd; ++d) atomicAdd(&sums[k * Dd + d], hrow[d]);
    }
    __syncthreads();
    for (int i = tid; i < KC * Dd; i += blockDim.x) sums[i] /= fmaxf(cnt[i >> 6], 1.0f);
    __syncthreads();
    // pass 2: squared deviation from cluster mean
    for (int r = tid; r < NPG; r += blockDim.x) {
        int n = b * NPG + r, k = asg[r];
        const float* hrow = H + (size_t)n * Dd;
        const float* mrow = &sums[k * Dd];
        float sq = 0.0f;
#pragma unroll
        for (int d = 0; d < Dd; ++d) { float t = hrow[d] - mrow[d]; sq += t * t; }
        atomicAdd(&fds[k], sq);
    }
    __syncthreads();
    if (tid == 0) {
        float fl = 0.0f;
        for (int k = 0; k < KC; ++k)
            if (cnt[k] > 0.0f) fl += (fds[k] / (float)Dd) / fmaxf(cnt[k], 1.0f);
        atomicAdd(&ws[OFF_SC + 3], fl);
    }
    if (tid < KC * KC) {
        int i = tid / KC, j = tid % KC;
        float pd = 0.0f;
#pragma unroll
        for (int d = 0; d < Dd; ++d) { float t = sums[i * Dd + d] - sums[j * Dd + d]; pd += t * t; }
        atomicAdd(&cgacc, pd / (float)Dd);
    }
    __syncthreads();
    if (tid == 0) ws[OFF_CG + b] = 0.5f * cgacc;
}

// ---------------- sequential scan over c_g (matches lax.scan) ----------------
__global__ void k_scan(float* ws) {
    float c = 0.0f;
    for (int b = 0; b < Bg; ++b) c = (c - ws[OFF_CG + b]) / (float)(KC - 1);
    ws[OFF_SC + 4] = c;
}

// ---------------- finalize: ortho + prototype terms + combine ----------------
__global__ void k_final(float* ws, float* out) {
    __shared__ float p[KC * Dd];
    if (threadIdx.x != 0) return;
    float oss = 0.0f;
    for (int i = 0; i < KC; ++i)
        for (int j = 0; j < KC; ++j) {
            float v = ws[OFF_SSB + i * 16 + j] - (i == j ? (float)Bg : 0.0f);
            oss += v * v;
        }
    float con = ws[OFF_SC + 2] + sqrtf(oss);
    for (int k = 0; k < KC; ++k) {
        float cdiv = ws[OFF_QCNT + k] + 0.1f;
        float nrm = 0.0f;
        for (int d = 0; d < Dd; ++d) {
            float t = ws[OFF_PRO + k * Dd + d] / cdiv;
            p[k * Dd + d] = t;
            nrm += t * t;
        }
        nrm = sqrtf(nrm) + 1e-15f;
        for (int d = 0; d < Dd; ++d) p[k * Dd + d] /= nrm;
    }
    float pdsum = 0.0f;
    for (int i = 0; i < KC; ++i)
        for (int j = 0; j < KC; ++j) {
            float pd = 0.0f;
            for (int d = 0; d < Dd; ++d) { float t = p[i * Dd + d] - p[j * Dd + d]; pd += t * t; }
            pdsum += pd / (float)Dd;
        }
    float center_loss = -0.5f * pdsum / ((float)(KC * (KC - 1)) * 0.5f);
    float l1p = 0.0f;
    for (int k = 0; k < KC; ++k) l1p += ws[OFF_QMIN + k];
    l1p /= (float)KC;
    float proto_term = 0.1f * (l1p + center_loss);                    // LAMBDA_PROTO
    float fea_term   = (ws[OFF_SC + 3] + ws[OFF_SC + 4]) / (float)Bg; // LAMBDA_FEA=1
    float con_term   = con / (float)Bg;                               // LAMBDA_CON=1
    out[0] = ws[OFF_SC + 0] + ws[OFF_SC + 1] + con_term + fea_term + proto_term;
}

extern "C" void kernel_launch(void* const* d_in, const int* in_sizes, int n_in,
                              void* d_out, int out_size, void* d_ws, size_t ws_size,
                              hipStream_t stream) {
    const float* Q  = (const float*)d_in[0];
    const float* E  = (const float*)d_in[1];
    const unsigned char* ind = (const unsigned char*)d_in[2];
    const float* S  = (const float*)d_in[3];
    const float* H  = (const float*)d_in[4];
    const int*   Lr = (const int*)d_in[5];
    const int*   Lc = (const int*)d_in[6];
    const float* Lv = (const float*)d_in[7];
    // d_in[8] = batch (deterministic n>>9, unused)
    const float* p1 = (const float*)d_in[9];
    const float* p2 = (const float*)d_in[10];
    const int* labels = (const int*)d_in[11];
    float* out = (float*)d_out;
    float* ws  = (float*)d_ws;
    float* den = ws + OFF_DEN;
    float* ssb = ws + OFF_SSB;
    float* pro = ws + OFF_PRO;
    float* Sn  = ws + OFF_SN;

    k_init<<<6, 256, 0, stream>>>(ws);
    k_ce<<<1, 256, 0, stream>>>(p1, p2, labels, ind, ws);
    k_denom<<<Bg, 256, 0, stream>>>(S, den);
    k_snorm<<<(Nn + 255) / 256, 256, 0, stream>>>(S, den, Sn);
    k_losssp<<<NNZ / 256, 256, 0, stream>>>(Sn, Lr, Lc, Lv, ws);
    k_ss_wmma<<<64, 256, 0, stream>>>(Sn, ssb);
    k_proto_wmma<<<128, 128, 0, stream>>>(Q, E, pro);
    k_qstats<<<128, 256, 0, stream>>>(Q, ws);
    k_feature<<<Bg, 256, 0, stream>>>(S, H, ws);
    k_scan<<<1, 1, 0, stream>>>(ws);
    k_final<<<1, 32, 0, stream>>>(ws, out);
}
